// MultiHeadedSelfAttention_7103875908242
// MI455X (gfx1250) — matrix operand order repaired and checked by reading, not compiled, not run on hardware
//
#include <hip/hip_runtime.h>

// ---------------------------------------------------------------------------
// MultiHeadedSelfAttention for MI455X (gfx1250), bf16 WMMA everywhere.
//   B=4, S=2048, E=1024, H=2, HD=512.  fp32 in/out, bf16 compute, f32 accum.
// Pipeline:
//   0) convert embedded f32 -> bf16;  transpose+convert Wq/Wk/Wv -> [H,HD,E] bf16
//   1) proj_gemm (WMMA, software-pipelined): Q[B,H,S,HD], K[B,H,S,HD], Vt[B,H,HD,S]
//   2) attn_kernel (WMMA flash attention, TDM-staged Q tile, online softmax)
// Workspace: ~73.4 MB carved from d_ws.
// ---------------------------------------------------------------------------

typedef __bf16 bf16;
typedef __attribute__((ext_vector_type(8)))  __bf16 v8bf;
typedef __attribute__((ext_vector_type(16))) __bf16 v16bf;
typedef __attribute__((ext_vector_type(8)))  float  v8f;

typedef unsigned int uint32x4 __attribute__((ext_vector_type(4)));
typedef int          int32x4  __attribute__((ext_vector_type(4)));
typedef int          int32x8  __attribute__((ext_vector_type(8)));

union AFrag { v16bf v; v8bf h[2]; };

#if __has_builtin(__builtin_amdgcn_tensor_load_to_lds)
#define HAVE_TDM 1
#else
#define HAVE_TDM 0
#endif

constexpr int NB = 4;     // batch
constexpr int NS = 2048;  // sequence
constexpr int NE = 1024;  // embed dim
constexpr int NH = 2;     // heads
constexpr int ND = 512;   // head dim

#define WMMA_BF16(A, Bf, C) \
  __builtin_amdgcn_wmma_f32_16x16x32_bf16(false, (A), false, (Bf), (short)0, (C), false, false)

// ---------------------------------------------------------------------------
// Prep kernels
// ---------------------------------------------------------------------------
__global__ __launch_bounds__(256) void cvt_f32_bf16(const float* __restrict__ in,
                                                    bf16* __restrict__ out, int n) {
  int i = blockIdx.x * blockDim.x + threadIdx.x;
  int stride = gridDim.x * blockDim.x;
  for (; i < n; i += stride) out[i] = (bf16)in[i];
}

// W: [NH][NE][ND] f32  ->  Wt: [NH][ND][NE] bf16
__global__ __launch_bounds__(256) void transpose_w(const float* __restrict__ W,
                                                   bf16* __restrict__ Wt) {
  const int n = NH * NE * ND;
  int i = blockIdx.x * blockDim.x + threadIdx.x;
  int stride = gridDim.x * blockDim.x;
  for (; i < n; i += stride) {
    int h = i / (NE * ND);
    int r = i - h * (NE * ND);
    int d = r / NE;
    int e = r - d * NE;
    Wt[i] = (bf16)W[(size_t)h * NE * ND + (size_t)e * ND + d];
  }
}

// ---------------------------------------------------------------------------
// Generic WMMA GEMM:  O[M,N] = A[M,NE] * B'[N,NE]^T   (inner dim = NE)
// Software-pipelined: next iteration's fragments are loaded before current
// WMMAs so s_wait_loadcnt overlaps with math.
// Block = 256 threads = 8 waves; wave -> 16x64 tile; workgroup -> 16 x 512.
// ---------------------------------------------------------------------------
__global__ __launch_bounds__(256)
void proj_gemm(const bf16* __restrict__ A, const bf16* __restrict__ Bm,
               bf16* __restrict__ O,
               long sAb, long sAh, long sBb, long sBh, long sOb, long sOh,
               int ldO) {
  const int t = threadIdx.x;
  const int lane = t & 31;
  const int w = t >> 5;
  const int z = blockIdx.z;
  const int b = z / NH, h = z % NH;
  A  += (size_t)b * sAb + (size_t)h * sAh;
  Bm += (size_t)b * sBb + (size_t)h * sBh;
  O  += (size_t)b * sOb + (size_t)h * sOh;

  const int m0 = blockIdx.x * 16;
  const int n0 = blockIdx.y * 512 + w * 64;
  const int cn = lane & 15;          // column / row-in-fragment index
  const int rb = (lane >> 4) * 8;    // K-offset half select & C row base

  v8f acc[4];
#pragma unroll
  for (int f = 0; f < 4; ++f) acc[f] = {};

  const bf16* ap  = A  + (size_t)(m0 + cn) * NE + (lane >> 4) * 16;
  const bf16* bp0 = Bm + (size_t)(n0 + cn) * NE + (lane >> 4) * 16;

  AFrag a_c;
  v16bf b_c[4];
  a_c.h[0] = *(const v8bf*)(ap);
  a_c.h[1] = *(const v8bf*)(ap + 8);
#pragma unroll
  for (int f = 0; f < 4; ++f) b_c[f] = *(const v16bf*)(bp0 + (size_t)f * 16 * NE);

#pragma unroll 4
  for (int e = 0; e < NE - 32; e += 32) {
    AFrag a_n;
    v16bf b_n[4];
    a_n.h[0] = *(const v8bf*)(ap + e + 32);
    a_n.h[1] = *(const v8bf*)(ap + e + 40);
#pragma unroll
    for (int f = 0; f < 4; ++f)
      b_n[f] = *(const v16bf*)(bp0 + (size_t)f * 16 * NE + e + 32);
#pragma unroll
    for (int f = 0; f < 4; ++f) acc[f] = WMMA_BF16(a_c.v, b_c[f], acc[f]);
    a_c = a_n;
#pragma unroll
    for (int f = 0; f < 4; ++f) b_c[f] = b_n[f];
  }
#pragma unroll
  for (int f = 0; f < 4; ++f) acc[f] = WMMA_BF16(a_c.v, b_c[f], acc[f]);

#pragma unroll
  for (int f = 0; f < 4; ++f)
#pragma unroll
    for (int i = 0; i < 8; ++i)
      O[(size_t)(m0 + rb + i) * ldO + (n0 + f * 16 + cn)] = (bf16)acc[f][i];
}

// ---------------------------------------------------------------------------
// Flash attention. Grid: (S/16, H, B). Block: 256 threads = 8 waves.
// Workgroup: 16 q-rows, full HD=512. k in blocks of 256: wave w computes
// scores for its 32 k-cols (full 512-deep reduction), cooperative online
// softmax via LDS, then wave w accumulates its 64-wide d-slice of ctx.
// Q tile is staged into LDS by the Tensor Data Mover (TDM) with row padding.
// ---------------------------------------------------------------------------
__global__ __launch_bounds__(256)
void attn_kernel(const bf16* __restrict__ Q, const bf16* __restrict__ K,
                 const bf16* __restrict__ Vt, float* __restrict__ Out) {
  __shared__ bf16  Qs[16][520];    // q tile, 8-elem row pad (TDM pad_amount)
  __shared__ float Ssc[16][260];   // raw scores for current 256-k block
  __shared__ bf16  Ps[16][264];    // probabilities (bf16) for P x V
  __shared__ float red[16][16];    // row-reduction partials
  __shared__ float mrow[16], lrow[16], nmrow[16], cscale[16];

  const int t = threadIdx.x;
  const int lane = t & 31;
  const int w = t >> 5;
  const int b = blockIdx.z, h = blockIdx.y;
  const int q0 = blockIdx.x * 16;

  const bf16* Qbase = Q  + (size_t)(b * NH + h) * NS * ND;
  const bf16* Kbase = K  + (size_t)(b * NH + h) * NS * ND;
  const bf16* Vbase = Vt + (size_t)(b * NH + h) * ND * NS;
  float* Obase = Out + (size_t)b * NS * (NH * ND) + (size_t)h * ND;

#if HAVE_TDM
  // ---- stage Q tile (16 x 512 bf16) via Tensor Data Mover ----
  // 2D descriptor: data_size=2B, tile=[512,16], row stride 512, LDS padding
  // of 4 DWORDs (16B) every 256 DWORDs (one row) -> matches Qs[16][520].
  if (w == 0) {
    unsigned long long ga =
        (unsigned long long)(const void*)(Qbase + (size_t)q0 * ND);
    unsigned int lds = (unsigned int)(size_t)(void*)&Qs[0][0];  // LDS offset
    uint32x4 g0;
    g0[0] = 1u;                                       // count=1 (valid user D#)
    g0[1] = lds;                                      // lds_addr
    g0[2] = (unsigned int)ga;                         // global_addr[31:0]
    g0[3] = (unsigned int)((ga >> 32) & 0x01FFFFFFu)  // global_addr[56:32]
            | 0x80000000u;                            // type=2 ("image")
    int32x8 g1;
    g1[0] = (1 << 16)      // data_size = 2 bytes
          | (1 << 20)      // pad_enable
          | (7 << 22)      // pad_interval = 256 DWORDs (one 512-elem row)
          | (3 << 25);     // pad_amount = 4 DWORDs (16B -> stride 520)
    g1[1] = (int)(512u << 16);   // tensor_dim0 = 512 (bits [79:48])
    g1[2] = (int)(16u << 16);    // tensor_dim1 = 16  (bits [111:80])
    g1[3] = (int)(512u << 16);   // tile_dim0 = 512   (bits [127:112])
    g1[4] = 16;                  // tile_dim1 = 16    (bits [143:128])
    g1[5] = 512;                 // tensor_dim0_stride = 512 (bits [207:160])
    g1[6] = 0;
    g1[7] = 0;
    int32x4 gz = {0, 0, 0, 0};
#if defined(__clang_major__) && (__clang_major__ >= 23)
    int32x8 gz8 = {0, 0, 0, 0, 0, 0, 0, 0};
    __builtin_amdgcn_tensor_load_to_lds(g0, g1, gz, gz, gz8, 0);
#else
    __builtin_amdgcn_tensor_load_to_lds(g0, g1, gz, gz, 0);
#endif
    __builtin_amdgcn_s_wait_tensorcnt(0);
  }
#else
  // ---- fallback: manual staged copy, 256 threads x 32 elements ----
  {
    int row = t >> 4;
    int col = (t & 15) * 32;
    const bf16* src = Qbase + (size_t)(q0 + row) * ND + col;
#pragma unroll
    for (int j = 0; j < 4; ++j)
      *(v8bf*)&Qs[row][col + j * 8] = *(const v8bf*)(src + j * 8);
  }
#endif
  if (t < 16) { mrow[t] = -3.0e38f; lrow[t] = 0.0f; }
  __syncthreads();

  v8f acc[4];
#pragma unroll
  for (int f = 0; f < 4; ++f) acc[f] = {};

  const int cn = lane & 15;
  const int rb = (lane >> 4) * 8;
  const float scale = 0.04419417382415922f;  // 1/sqrt(512)

  const int nkb = q0 / 256 + 1;  // causal: only blocks with k0 <= q0+15
  for (int kb = 0; kb < nkb; ++kb) {
    const int k0 = kb * 256;

    // ---- scores: wave w covers k-cols [k0 + 32w, k0 + 32w + 32) ----
    v8f sc0 = {}, sc1 = {};
    {
      const bf16* kp0 = Kbase + (size_t)(k0 + w * 32 + cn) * ND + (lane >> 4) * 16;
      const bf16* kp1 = kp0 + (size_t)16 * ND;
      if (kb + 1 < nkb)
        __builtin_prefetch(Kbase + (size_t)(k0 + 256 + w * 32 + cn) * ND, 0, 1);
      AFrag a_c;
      v16bf b0_c, b1_c;
      a_c.h[0] = *(const v8bf*)&Qs[cn][2 * rb];
      a_c.h[1] = *(const v8bf*)&Qs[cn][8 + 2 * rb];
      b0_c = *(const v16bf*)(kp0);
      b1_c = *(const v16bf*)(kp1);
#pragma unroll 5
      for (int e = 0; e < ND - 32; e += 32) {
        AFrag a_n;
        v16bf b0_n, b1_n;
        a_n.h[0] = *(const v8bf*)&Qs[cn][e + 32 + 2 * rb];
        a_n.h[1] = *(const v8bf*)&Qs[cn][e + 40 + 2 * rb];
        b0_n = *(const v16bf*)(kp0 + e + 32);
        b1_n = *(const v16bf*)(kp1 + e + 32);
        sc0 = WMMA_BF16(a_c.v, b0_c, sc0);
        sc1 = WMMA_BF16(a_c.v, b1_c, sc1);
        a_c = a_n; b0_c = b0_n; b1_c = b1_n;
      }
      sc0 = WMMA_BF16(a_c.v, b0_c, sc0);
      sc1 = WMMA_BF16(a_c.v, b1_c, sc1);
    }
    // scale + causal mask + spill to LDS
    {
      int col0 = k0 + w * 32 + cn;
#pragma unroll
      for (int i = 0; i < 8; ++i) {
        int row = q0 + rb + i;
        float v0 = sc0[i] * scale;
        float v1 = sc1[i] * scale;
        Ssc[rb + i][w * 32 + cn]      = (col0 <= row)      ? v0 : -1.0e30f;
        Ssc[rb + i][w * 32 + 16 + cn] = (col0 + 16 <= row) ? v1 : -1.0e30f;
      }
    }
    __syncthreads();  // B1: scores complete

    // ---- cooperative online softmax over 16 x 256 ----
    const int r = t >> 4, seg = t & 15;
    {
      float lm = -3.0e38f;
#pragma unroll
      for (int c = 0; c < 16; ++c) lm = fmaxf(lm, Ssc[r][seg * 16 + c]);
      red[r][seg] = lm;
    }
    __syncthreads();  // B2
    if (t < 16) {
      float bm = -3.0e38f;
#pragma unroll
      for (int c = 0; c < 16; ++c) bm = fmaxf(bm, red[t][c]);
      float nm = fmaxf(mrow[t], bm);
      nmrow[t] = nm;
      cscale[t] = __expf(mrow[t] - nm);
    }
    __syncthreads();  // B3
    {
      float nm = nmrow[r];
      float s = 0.0f;
#pragma unroll
      for (int c = 0; c < 16; ++c) {
        float p = __expf(Ssc[r][seg * 16 + c] - nm);
        Ps[r][seg * 16 + c] = (bf16)p;
        s += p;
      }
      red[r][seg] = s;
    }
    __syncthreads();  // B4
    if (t < 16) {
      float bs = 0.0f;
#pragma unroll
      for (int c = 0; c < 16; ++c) bs += red[t][c];
      lrow[t] = lrow[t] * cscale[t] + bs;
      mrow[t] = nmrow[t];
    }
    __syncthreads();  // B5: Ps + cscale ready

    // ---- rescale running ctx ----
    float cs[8];
#pragma unroll
    for (int i = 0; i < 8; ++i) cs[i] = cscale[rb + i];
#pragma unroll
    for (int f = 0; f < 4; ++f)
#pragma unroll
      for (int i = 0; i < 8; ++i) acc[f][i] *= cs[i];

    // ---- ctx += P(16x256) x V(256x64)  (wave's d-slice = [64w, 64w+64)) ----
    {
      const bf16* vb = Vbase + k0 + (lane >> 4) * 16;
      AFrag a_c;
      v16bf bv_c[4];
      a_c.h[0] = *(const v8bf*)&Ps[cn][2 * rb];
      a_c.h[1] = *(const v8bf*)&Ps[cn][8 + 2 * rb];
#pragma unroll
      for (int f = 0; f < 4; ++f)
        bv_c[f] = *(const v16bf*)(vb + (size_t)(w * 64 + f * 16 + cn) * NS);
#pragma unroll 3
      for (int kk = 0; kk < 224; kk += 32) {
        AFrag a_n;
        v16bf bv_n[4];
        a_n.h[0] = *(const v8bf*)&Ps[cn][kk + 32 + 2 * rb];
        a_n.h[1] = *(const v8bf*)&Ps[cn][kk + 40 + 2 * rb];
#pragma unroll
        for (int f = 0; f < 4; ++f)
          bv_n[f] = *(const v16bf*)(vb + (size_t)(w * 64 + f * 16 + cn) * NS + kk + 32);
#pragma unroll
        for (int f = 0; f < 4; ++f) acc[f] = WMMA_BF16(a_c.v, bv_c[f], acc[f]);
        a_c = a_n;
#pragma unroll
        for (int f = 0; f < 4; ++f) bv_c[f] = bv_n[f];
      }
#pragma unroll
      for (int f = 0; f < 4; ++f) acc[f] = WMMA_BF16(a_c.v, bv_c[f], acc[f]);
    }
  }

  // ---- normalize and write fp32 output [B,S,H*HD] ----
  float li[8];
#pragma unroll
  for (int i = 0; i < 8; ++i) li[i] = 1.0f / lrow[rb + i];
#pragma unroll
  for (int f = 0; f < 4; ++f)
#pragma unroll
    for (int i = 0; i < 8; ++i)
      Obase[(size_t)(q0 + rb + i) * (NH * ND) + (w * 64 + f * 16 + cn)] =
          acc[f][i] * li[i];
}

// ---------------------------------------------------------------------------
extern "C" void kernel_launch(void* const* d_in, const int* in_sizes, int n_in,
                              void* d_out, int out_size, void* d_ws,
                              size_t ws_size, hipStream_t stream) {
  (void)in_sizes; (void)n_in; (void)out_size; (void)ws_size;

  const float* emb = (const float*)d_in[0];
  const float* Wq  = (const float*)d_in[1];
  const float* Wk  = (const float*)d_in[2];
  const float* Wv  = (const float*)d_in[3];
  float* out = (float*)d_out;

  // workspace carve-out (~73.4 MB total)
  char* p = (char*)d_ws;
  bf16* embB = (bf16*)p; p += (size_t)NB * NS * NE * 2;
  bf16* Wqt  = (bf16*)p; p += (size_t)NH * ND * NE * 2;
  bf16* Wkt  = (bf16*)p; p += (size_t)NH * ND * NE * 2;
  bf16* Wvt  = (bf16*)p; p += (size_t)NH * ND * NE * 2;
  bf16* Qb   = (bf16*)p; p += (size_t)NB * NH * NS * ND * 2;
  bf16* Kb   = (bf16*)p; p += (size_t)NB * NH * NS * ND * 2;
  bf16* Vt   = (bf16*)p; p += (size_t)NB * NH * ND * NS * 2;

  // 0) conversions
  cvt_f32_bf16<<<4096, 256, 0, stream>>>(emb, embB, NB * NS * NE);
  transpose_w<<<1024, 256, 0, stream>>>(Wq, Wqt);
  transpose_w<<<1024, 256, 0, stream>>>(Wk, Wkt);
  transpose_w<<<1024, 256, 0, stream>>>(Wv, Wvt);

  // 1) projections
  // Q/K: O[s,d] = emb[b][s,:] . Wt[h][d,:]   (M=S, N=HD)
  dim3 gqk(NS / 16, 1, NB * NH);
  proj_gemm<<<gqk, 256, 0, stream>>>(embB, Wqt, Qb,
      (long)NS * NE, 0, 0, (long)ND * NE,
      (long)NH * NS * ND, (long)NS * ND, ND);
  proj_gemm<<<gqk, 256, 0, stream>>>(embB, Wkt, Kb,
      (long)NS * NE, 0, 0, (long)ND * NE,
      (long)NH * NS * ND, (long)NS * ND, ND);
  // Vt: O[d,s] = Wt[h][d,:] . emb[b][s,:]    (M=HD, N=S) -> V transposed
  dim3 gv(ND / 16, NS / 512, NB * NH);
  proj_gemm<<<gv, 256, 0, stream>>>(Wvt, embB, Vt,
      0, (long)ND * NE, (long)NS * NE, 0,
      (long)NH * ND * NS, (long)ND * NS, NS);

  // 2) attention
  dim3 ga(NS / 16, NH, NB);
  attn_kernel<<<ga, 256, 0, stream>>>(Qb, Kb, Vt, out);
}
